// MSPELoss_3942779978267
// MI455X (gfx1250) — matrix operand order repaired and checked by reading, not compiled, not run on hardware
//
#include <hip/hip_runtime.h>

typedef __attribute__((ext_vector_type(16))) _Float16 v16h;
typedef __attribute__((ext_vector_type(8)))  float    v8f;

#define MM    5
#define NPERM 120

// Compile-time B tiles in WMMA 32x16 f16 register layout, as raw f16 bits.
// BT.h[n][L][t] is element t of lane L for N-tile n:
//   column p = n*16 + (L&15), K = (L>=16 ? 16 : 0) + t,
//   value = 1.0h iff lexicographic permutation p maps position j=K%5 to
//   index i=K/5 (K<25), matching itertools.permutations(range(5)) order.
// Pad columns (p>=120) and pad K (25..31) are zero.
struct BTab {
    alignas(32) unsigned short h[8][32][16];
    constexpr BTab() : h{} {
        for (int p = 0; p < 128; ++p) {
            unsigned mask = 0u;
            if (p < NPERM) {
                int pool[5] = {0, 1, 2, 3, 4};
                const int fact[4] = {24, 6, 2, 1};
                int rem = p;
                for (int j = 0; j < 5; ++j) {
                    int dsel = 0;
                    if (j < 4) { dsel = rem / fact[j]; rem -= dsel * fact[j]; }
                    const int val = pool[dsel];                 // = perm[j]
                    mask |= 1u << (val * 5 + j);
                    for (int t = dsel; t < 4; ++t) pool[t] = pool[t + 1];
                }
            }
            const int n = p >> 4;       // tile index
            const int c = p & 15;       // column-within-tile = lane & 15
            for (int half = 0; half < 2; ++half) {
                const int L = half * 16 + c;
                for (int t = 0; t < 16; ++t) {
                    const int K = half * 16 + t;
                    h[n][L][t] = ((mask >> K) & 1u) ? (unsigned short)0x3C00
                                                    : (unsigned short)0x0000;
                }
            }
        }
    }
};
__device__ __constant__ BTab BT;

__device__ __forceinline__ float mod_err2(float x) {
    const float PI_F      = 3.14159265358979323846f;
    const float HALF_PI_F = 1.57079632679489661923f;
    const float INV_PI_F  = 0.31830988618379067154f;
    float y = x + HALF_PI_F;
    float q = floorf(y * INV_PI_F);
    float m = y - q * PI_F;        // floor-mod -> [0, pi)
    float e = m - HALF_PI_F;
    return e * e;
}

// One wave = 16 batch rows x 120 perms via 8x v_wmma_f32_16x16x32_f16.
__global__ void __launch_bounds__(256) mspe_wmma_kernel(
    const float* __restrict__ pred, const float* __restrict__ doa,
    float* __restrict__ partial, int B)
{
    __shared__ float lds[8];
    const int lane    = threadIdx.x & 31;
    const int wave    = threadIdx.x >> 5;
    const int hi      = lane >> 4;                    // 0: lanes 0-15, 1: lanes 16-31
    const int rowBase = blockIdx.x * 128 + wave * 16;
    const int myRow   = rowBase + (lane & 15);

    // ---- load the 8 pre-laid-out B tiles early (L2-resident 8KB, coalesced) ----
    v16h btile[8];
    #pragma unroll
    for (int n = 0; n < 8; ++n)
        btile[n] = *reinterpret_cast<const v16h*>(&BT.h[n][lane][0]);

    // ---- load one row of pred/doa (zeros OOB -> contributes 0 to the sum) ----
    float p[MM], d[MM];
    if (myRow < B) {
        const float* pp = pred + (size_t)myRow * MM;
        const float* dd = doa  + (size_t)myRow * MM;
        #pragma unroll
        for (int t = 0; t < MM; ++t) { p[t] = pp[t]; d[t] = dd[t]; }
    } else {
        #pragma unroll
        for (int t = 0; t < MM; ++t) { p[t] = 0.0f; d[t] = 0.0f; }
    }

    // ---- 25-entry pairwise squared-error table (constant indices only) ----
    float e2[MM * MM];
    #pragma unroll
    for (int i = 0; i < MM; ++i)
        #pragma unroll
        for (int j = 0; j < MM; ++j)
            e2[i * MM + j] = mod_err2(p[i] - d[j]);

    // ---- pack A tile: 16x32 f16, contiguous K = 16*hi + t (K padded 25->32 with zeros) ----
    v16h a;
    #pragma unroll
    for (int t = 0; t < 16; ++t) {
        const int k0 = t;                         // K for lanes 0-15
        const int k1 = t + 16;                    // K for lanes 16-31
        const float v0 = (k0 < MM * MM) ? e2[k0] : 0.0f;   // compile-time index
        const float v1 = (k1 < MM * MM) ? e2[k1] : 0.0f;
        a[t] = (_Float16)(hi ? v1 : v0);
    }

    // ---- 8 N-tiles of 16 permutation columns each ----
    v8f acc[8];
    #pragma unroll
    for (int n = 0; n < 8; ++n) {
        v8f c = {0.f, 0.f, 0.f, 0.f, 0.f, 0.f, 0.f, 0.f};
        c = __builtin_amdgcn_wmma_f32_16x16x32_f16(
                /*neg_a=*/false, a, /*neg_b=*/false, btile[n],
                /*c_mod=*/(short)0, c, /*reuse_a=*/false, /*reuse_b=*/false);
        acc[n] = c;
    }

    // ---- min over 120 permutation columns ----
    // lane holds column N = lane&15 of each tile; component r is row M = 8*hi + r
    float mn[8];
    #pragma unroll
    for (int r = 0; r < 8; ++r) mn[r] = 3.0e38f;
    #pragma unroll
    for (int n = 0; n < 8; ++n) {
        const int  pcol  = n * 16 + (lane & 15);
        const bool valid = (pcol < NPERM);        // exclude all-zero pad columns
        #pragma unroll
        for (int r = 0; r < 8; ++r) {
            const float v = valid ? acc[n][r] : 3.0e38f;
            mn[r] = fminf(mn[r], v);
        }
    }
    // min-tree across the 16 lanes of each half (xor bits 0..3 only)
    #pragma unroll
    for (int off = 1; off < 16; off <<= 1) {
        #pragma unroll
        for (int r = 0; r < 8; ++r) {
            const float o = __shfl_xor(mn[r], off, 32);
            mn[r] = fminf(mn[r], o);
        }
    }

    // per-half row sum; mspe = dot/5
    float s = 0.0f;
    #pragma unroll
    for (int r = 0; r < 8; ++r) s += mn[r];
    s *= 0.2f;

    // wave total = lanes(0-15 rows M0-7) + lanes(16-31 rows M8-15)
    const float other = __shfl_xor(s, 16, 32);
    const float wsum  = s + other;
    if (lane == 0) lds[wave] = wsum;
    __syncthreads();
    if (threadIdx.x == 0) {
        float t = 0.0f;
        #pragma unroll
        for (int w = 0; w < 8; ++w) t += lds[w];
        partial[blockIdx.x] = t;
    }
}

// Deterministic final reduction of per-block partials into d_out[0].
__global__ void __launch_bounds__(256) mspe_reduce_kernel(
    const float* __restrict__ partial, int n, float* __restrict__ out)
{
    __shared__ float lds[256];
    float s = 0.0f;
    for (int i = threadIdx.x; i < n; i += 256) s += partial[i];
    lds[threadIdx.x] = s;
    __syncthreads();
    #pragma unroll
    for (int off = 128; off > 0; off >>= 1) {
        if ((int)threadIdx.x < off) lds[threadIdx.x] += lds[threadIdx.x + off];
        __syncthreads();
    }
    if (threadIdx.x == 0) out[0] = lds[0];
}

extern "C" void kernel_launch(void* const* d_in, const int* in_sizes, int n_in,
                              void* d_out, int out_size, void* d_ws, size_t ws_size,
                              hipStream_t stream) {
    const float* pred = (const float*)d_in[0];
    const float* doa  = (const float*)d_in[1];
    float* out     = (float*)d_out;
    float* partial = (float*)d_ws;

    const int B      = in_sizes[0] / MM;         // 65536
    const int blocks = (B + 127) / 128;          // 16 rows/wave * 8 waves

    mspe_wmma_kernel<<<blocks, 256, 0, stream>>>(pred, doa, partial, B);
    mspe_reduce_kernel<<<1, 256, 0, stream>>>(partial, blocks, out);
}
